// SparseLinear_82935818485772
// MI455X (gfx1250) — matrix operand order repaired and checked by reading, not compiled, not run on hardware
//
#include <hip/hip_runtime.h>
#include <hip/hip_bf16.h>
#include <stdint.h>

// ---------------------------------------------------------------------------
// SparseLinear: y = x @ W.T + bias, W given as fixed COO (10% dense).
// Strategy: densify W into bf16 *pre-swizzled WMMA B-fragment layout* in d_ws,
// then run a dense bf16 WMMA GEMM (f32 accumulate) with 256x256 block tiles,
// LDS staging, and fused bias epilogue.  Everything L2-resident on MI455X.
// ---------------------------------------------------------------------------

typedef __attribute__((ext_vector_type(16))) __bf16 v16bf;
typedef __attribute__((ext_vector_type(8)))  float  v8f;

#define IN_F    4096
#define OUT_F   4096
#define BATCH   8192
#define NT      (OUT_F / 16)   // 256 n-tiles of 16
#define KT      (IN_F / 32)    // 128 k-steps of 32
#define BM      256
#define BN      256
#define THREADS 512            // 16 waves of 32

__device__ __forceinline__ unsigned short f2bf(float f) {
  unsigned int u = __float_as_uint(f);
  u += 0x7FFFu + ((u >> 16) & 1u);          // round-to-nearest-even
  return (unsigned short)(u >> 16);
}

// wbuf layout (ushort/bf16): [kTile][nTile][lane(32)][e(16)]
//   lane = 16*khalf + (o & 15), khalf = (c>>4)&1, e = c & 15
// This is the contiguous k order (k = 16*khalf + e), the same order the A fragments use,
// so the GEMM loads fragments as contiguous 32B per lane.

__global__ void zero_kernel(uint4* __restrict__ p, int n16) {
  int i = blockIdx.x * blockDim.x + threadIdx.x;
  if (i < n16) p[i] = make_uint4(0u, 0u, 0u, 0u);
}

__global__ void scatter_kernel(const int* __restrict__ rows,
                               const int* __restrict__ cols,
                               const float* __restrict__ vals,
                               unsigned short* __restrict__ wbuf, int nnz) {
  int i = blockIdx.x * blockDim.x + threadIdx.x;
  if (i >= nnz) return;
  int o = rows[i];              // out-feature (N)
  int c = cols[i];              // in-feature  (K)
  int kTile = c >> 5;
  int khalf = (c >> 4) & 1;
  int e     = c & 15;
  int nTile = o >> 4;
  int lane  = khalf * 16 + (o & 15);
  size_t idx = ((((size_t)kTile * NT + nTile) * 32) + lane) * 16 + e;
  wbuf[idx] = f2bf(vals[i]);
}

__global__ __launch_bounds__(THREADS)
void spmm_wmma_kernel(const float* __restrict__ x,
                      const unsigned short* __restrict__ wbuf,
                      const float* __restrict__ bias,
                      float* __restrict__ out) {
  // LDS tiles, already in fragment layout: [tile(16)][lane(32)][e(16)] bf16
  __shared__ __align__(32) unsigned short lA[16 * 32 * 16];  // 16 KB
  __shared__ __align__(32) unsigned short lB[16 * 32 * 16];  // 16 KB

  const int tid  = threadIdx.x;
  const int lane = tid & 31;
  const int wave = tid >> 5;          // 0..15
  const int rw   = wave >> 2;         // wave row 0..3  (M)
  const int cw   = wave & 3;          // wave col 0..3  (N)

  const int mBase  = blockIdx.y * BM;          // batch-row base
  const int ntBase = blockIdx.x * (BN / 16);   // n-tile base (16 tiles/block)

  // Cooperative-load mapping: thread -> one 32B fragment slice.
  const int ctile = tid >> 5;         // which of 16 tiles this thread fills
  const int cm    = lane & 15;        // A: row-in-tile
  const int ckh   = lane >> 4;        // A: k-half select
  const float* xsrc = x + (size_t)(mBase + ctile * 16 + cm) * IN_F + ckh * 16;
  const uint4* bsrc = (const uint4*)wbuf;

  v8f acc[4][4] = {};

  for (int kt = 0; kt < KT; ++kt) {
    // ---- global -> registers (A converted fp32->bf16; B straight copy) ----
    const float* s = xsrc + kt * 32;
    float4 f0 = *(const float4*)(s);
    float4 f1 = *(const float4*)(s + 4);
    float4 f2 = *(const float4*)(s + 8);
    float4 f3 = *(const float4*)(s + 12);

    size_t boff = ((((size_t)kt * NT + ntBase + ctile) * 32) + lane) * 2; // uint4 units
    uint4 b0 = bsrc[boff];
    uint4 b1 = bsrc[boff + 1];

    if (kt + 1 < KT) {
      __builtin_prefetch(xsrc + (kt + 1) * 32, 0, 0);
      __builtin_prefetch(bsrc + boff + (size_t)NT * 64, 0, 0);
    }

    __syncthreads();   // previous iteration done reading LDS

    // Scalar ushort stores: the vectorizer merges these into ds_store_b128
    // with plain v_lshl/v_or packing (leaner than manual dword assembly).
    unsigned short* pa = &lA[tid * 16];
    pa[0]  = f2bf(f0.x); pa[1]  = f2bf(f0.y); pa[2]  = f2bf(f0.z); pa[3]  = f2bf(f0.w);
    pa[4]  = f2bf(f1.x); pa[5]  = f2bf(f1.y); pa[6]  = f2bf(f1.z); pa[7]  = f2bf(f1.w);
    pa[8]  = f2bf(f2.x); pa[9]  = f2bf(f2.y); pa[10] = f2bf(f2.z); pa[11] = f2bf(f2.w);
    pa[12] = f2bf(f3.x); pa[13] = f2bf(f3.y); pa[14] = f2bf(f3.z); pa[15] = f2bf(f3.w);
    *(uint4*)&lB[tid * 16]     = b0;
    *(uint4*)&lB[tid * 16 + 8] = b1;

    __syncthreads();   // tiles visible

    // ---- fragments + 16 WMMAs per wave per k-step ----
    v16bf a[4], b[4];
    #pragma unroll
    for (int i = 0; i < 4; ++i)
      a[i] = *(const v16bf*)&lA[((rw * 4 + i) * 32 + lane) * 16];
    #pragma unroll
    for (int j = 0; j < 4; ++j)
      b[j] = *(const v16bf*)&lB[((cw * 4 + j) * 32 + lane) * 16];

    #pragma unroll
    for (int i = 0; i < 4; ++i) {
      #pragma unroll
      for (int j = 0; j < 4; ++j) {
        acc[i][j] = __builtin_amdgcn_wmma_f32_16x16x32_bf16(
            /*neg_a=*/false, a[i], /*neg_b=*/false, b[j],
            /*c_mod=*/(short)0, acc[i][j],
            /*reuse_a=*/false, /*reuse_b=*/false);
      }
    }
  }

  // ---- epilogue: C layout (VGPR r -> M = r + 8*(lane>=16), N = lane&15) ----
  const int n  = lane & 15;
  const int mo = (lane >> 4) * 8;
  #pragma unroll
  for (int j = 0; j < 4; ++j) {
    int col = (ntBase + cw * 4 + j) * 16 + n;
    float bv = bias[col];
    #pragma unroll
    for (int i = 0; i < 4; ++i) {
      int row0 = mBase + (rw * 4 + i) * 16 + mo;
      #pragma unroll
      for (int r = 0; r < 8; ++r)
        out[(size_t)(row0 + r) * OUT_F + col] = acc[i][j][r] + bv;
    }
  }
}

extern "C" void kernel_launch(void* const* d_in, const int* in_sizes, int n_in,
                              void* d_out, int out_size, void* d_ws, size_t ws_size,
                              hipStream_t stream) {
  const float* x       = (const float*)d_in[0];
  const int*   indices = (const int*)d_in[1];   // [2, NNZ]: rows then cols
  const float* values  = (const float*)d_in[2];
  const float* bias    = (const float*)d_in[3];
  float*       out     = (float*)d_out;
  const int    nnz     = in_sizes[2];

  unsigned short* wbuf = (unsigned short*)d_ws;  // needs OUT_F*IN_F*2 = 33.5 MB

  // 1) zero the densified-W fragment buffer
  const int n16 = (int)(((size_t)OUT_F * IN_F) / 8);  // uint4 count
  hipLaunchKernelGGL(zero_kernel, dim3((n16 + 255) / 256), dim3(256), 0, stream,
                     (uint4*)wbuf, n16);

  // 2) scatter COO values into WMMA B-fragment layout (bf16)
  hipLaunchKernelGGL(scatter_kernel, dim3((nnz + 255) / 256), dim3(256), 0, stream,
                     indices, indices + nnz, values, wbuf, nnz);

  // 3) dense bf16 WMMA GEMM with fused bias
  hipLaunchKernelGGL(spmm_wmma_kernel, dim3(OUT_F / BN, BATCH / BM), dim3(THREADS),
                     0, stream, x, wbuf, bias, out);
}
